// ProbAttention_15573551416052
// MI455X (gfx1250) — matrix operand order repaired and checked by reading, not compiled, not run on hardware
//
#include <hip/hip_runtime.h>
#include <hip/hip_bf16.h>

// ---------------- problem constants ----------------
#define B_   8
#define Lq   4096
#define Sk   4096
#define Dm   1024
#define Hh   16
#define Dk   64
#define Uu   45      // 5 * ceil(ln(4096)) = 45
#define QROWS 90     // per batch: 45 perm-sampled rows + rows 0..44

typedef __attribute__((ext_vector_type(16))) __bf16 v16bf;
typedef __attribute__((ext_vector_type(8)))  float  v8f;

// async-copy pointer types: <4 x i32> in global (AS1) / LDS (AS3) address spaces
typedef int v4i __attribute__((vector_size(16)));
typedef __attribute__((address_space(1))) v4i as1_v4i;
typedef __attribute__((address_space(3))) v4i as3_v4i;

#if defined(__gfx1250__) && __has_builtin(__builtin_amdgcn_global_load_async_to_lds_b128)
#define HAVE_ASYNC_LDS 1
#else
#define HAVE_ASYNC_LDS 0
#endif

union FragAB { v16bf v; unsigned int u[8]; };
union FragC  { v8f v; float f[8]; };
union Vec4U  { uint4 q; unsigned short s[8]; };

__device__ __forceinline__ unsigned short f32_to_bf16_bits(float x) {
  unsigned u = __float_as_uint(x);
  return (unsigned short)((u + 0x7FFFu + ((u >> 16) & 1u)) >> 16);  // RNE
}

// A-matrix 16x32 bf16 fragment (ISA 7.12.2): lane m=l&15; low half K={0..7,16..23}, high half +8.
__device__ __forceinline__ v16bf load_frag_a(const unsigned short* rowPtr, int lane) {
  FragAB f;
  const unsigned int* p32 = (const unsigned int*)rowPtr;  // row base 4B aligned
  int kb2 = (lane < 16) ? 0 : 4;                          // kb/2
#pragma unroll
  for (int vg = 0; vg < 4; ++vg) f.u[vg]     = p32[kb2 + vg];
#pragma unroll
  for (int vg = 0; vg < 4; ++vg) f.u[4 + vg] = p32[8 + kb2 + vg];
  return f.v;
}

// B-matrix 32x16 bf16 fragment: lane n=l&15; low half K={0..7,16..23}, high half +8.
__device__ __forceinline__ v16bf load_frag_b(const unsigned short* rowPtr, int lane) {
  FragAB f;
  const uint4* p = (const uint4*)(rowPtr + ((lane < 16) ? 0 : 8));  // 16B aligned
  uint4 a = p[0], b = p[2];
  f.u[0]=a.x; f.u[1]=a.y; f.u[2]=a.z; f.u[3]=a.w;
  f.u[4]=b.x; f.u[5]=b.y; f.u[6]=b.z; f.u[7]=b.w;
  return f.v;
}

__device__ __forceinline__ void wait_asynccnt0() {
#if HAVE_ASYNC_LDS
#if __has_builtin(__builtin_amdgcn_s_wait_asynccnt)
  __builtin_amdgcn_s_wait_asynccnt(0);
#else
  asm volatile("s_wait_asynccnt 0x0" ::: "memory");
#endif
#endif
}

// ---------------- f32 -> bf16 bulk convert ----------------
__global__ void cvt_bf16_kernel(const float* __restrict__ in, unsigned short* __restrict__ out, long n) {
  long i = (long)blockIdx.x * blockDim.x + threadIdx.x;
  long stride = (long)gridDim.x * blockDim.x;
  for (; i < n; i += stride) out[i] = f32_to_bf16_bits(in[i]);
}

// Gather the 90 query rows per batch that matter: rows perm[0..44], then rows 0..44.
__global__ void gather_q_kernel(const float* __restrict__ q, const int* __restrict__ perm,
                                unsigned short* __restrict__ out) {
  int r = blockIdx.x;             // B_*QROWS
  int b = r / QROWS, j = r % QROWS;
  int src = (j < Uu) ? perm[j] : (j - Uu);
  const float* pin = q + ((long)b * Lq + src) * Dm;
  unsigned short* po = out + (long)r * Dm;
  for (int i = threadIdx.x; i < Dm; i += blockDim.x) po[i] = f32_to_bf16_bits(pin[i]);
}

// ---------------- generic bf16 WMMA GEMM: C = A(MxK) * B(KxN) + bias ----------------
// Block tile 128x128x32, 8 waves, each wave owns a 64x32 sub-tile (4x2 wmma accumulators).
// A tile staged via GLOBAL_LOAD_ASYNC_TO_LDS_B128 (ASYNCcnt) when the tile is full;
// B tile staged via VGPRs with an in-LDS transpose (WMMA B operand wants [n][k]).
// z-batch addressing: off = (z/headMod)*strideZ + (z%headMod)*strideH  (headMod=1 -> linear).
#define BM 128
#define BN 128
#define BK 32

__global__ __launch_bounds__(256) void gemm_bf16_kernel(
    const unsigned short* __restrict__ A, const unsigned short* __restrict__ Bm,
    const float* __restrict__ bias,
    float* __restrict__ Cf, unsigned short* __restrict__ Cb,
    int M, int N, int K, int lda, int ldb, int ldc, int headMod,
    long sAz, long sAh, long sBz, long sBh, long sCz, long sCh) {
  __shared__ unsigned short lA[BM * BK];   // [m][k]
  __shared__ unsigned short lBt[BN * BK];  // transposed [n][k]

  int z  = blockIdx.z;
  int zo = z / headMod, zh = z % headMod;
  A  += zo * sAz + zh * sAh;
  Bm += zo * sBz + zh * sBh;
  long coff = zo * sCz + zh * sCh;

  int bm = blockIdx.y * BM, bn = blockIdx.x * BN;
  int tid = threadIdx.x, lane = tid & 31, wid = tid >> 5;
  int wm = (wid >> 2) * 64, wn = (wid & 3) * 32;
  bool fullA = (bm + BM <= M);

  FragC acc[4][2];
#pragma unroll
  for (int i = 0; i < 4; ++i)
#pragma unroll
    for (int j = 0; j < 2; ++j)
#pragma unroll
      for (int e = 0; e < 8; ++e) acc[i][j].f[e] = 0.f;

  for (int k0 = 0; k0 < K; k0 += BK) {
#if HAVE_ASYNC_LDS
    if (fullA) {
      // async DMA straight into LDS, no VGPR round-trip; tracked by ASYNCcnt
      for (int i = tid; i < (BM * BK / 8); i += 256) {
        int r = i >> 2, c = (i & 3) * 8;
        __builtin_amdgcn_global_load_async_to_lds_b128(
            (as1_v4i*)(void*)(A + (long)(bm + r) * lda + k0 + c),
            (as3_v4i*)(void*)&lA[r * BK + c], 0, 0);
      }
    } else
#else
    (void)fullA;
#endif
    {
      // guarded synchronous staging (partial M tiles; rows >= M zero-padded)
      for (int i = tid; i < (BM * BK / 8); i += 256) {
        int r = i >> 2, c = (i & 3) * 8, gr = bm + r;
        uint4 val; val.x = val.y = val.z = val.w = 0u;
        if (gr < M) val = *(const uint4*)(A + (long)gr * lda + k0 + c);
        *(uint4*)(&lA[r * BK + c]) = val;
      }
    }
    // stage B tile transposed into [n][k] (cols >= N zero-padded)
    for (int i = tid; i < (BN * BK / 8); i += 256) {
      int kr = i >> 4, c = (i & 15) * 8;
      Vec4U v; v.q.x = v.q.y = v.q.z = v.q.w = 0u;
      if (bn + c < N) v.q = *(const uint4*)(Bm + (long)(k0 + kr) * ldb + bn + c);
#pragma unroll
      for (int e = 0; e < 8; ++e) lBt[(c + e) * BK + kr] = v.s[e];
    }
#if HAVE_ASYNC_LDS
    if (fullA) wait_asynccnt0();
#endif
    __syncthreads();

    if (k0 + BK < K)  // global_prefetch_b8 for the next A tile
      __builtin_prefetch(A + (long)(bm + (tid >> 2)) * lda + k0 + BK, 0, 1);

    v16bf afr[4], bfr[2];
#pragma unroll
    for (int sm = 0; sm < 4; ++sm)
      afr[sm] = load_frag_a(&lA[(wm + sm * 16 + (lane & 15)) * BK], lane);
#pragma unroll
    for (int sn = 0; sn < 2; ++sn)
      bfr[sn] = load_frag_b(&lBt[(wn + sn * 16 + (lane & 15)) * BK], lane);
#pragma unroll
    for (int sm = 0; sm < 4; ++sm)
#pragma unroll
      for (int sn = 0; sn < 2; ++sn)
        acc[sm][sn].v = __builtin_amdgcn_wmma_f32_16x16x32_bf16(
            false, afr[sm], false, bfr[sn], (short)0, acc[sm][sn].v, false, false);
    __syncthreads();
  }

  // epilogue: C layout — lane n=l&15; VGPR v holds M=v (lanes 0-15) / M=v+8 (lanes 16-31)
  int nloc = lane & 15, mhalf = (lane >> 4) * 8;
#pragma unroll
  for (int sm = 0; sm < 4; ++sm)
#pragma unroll
    for (int sn = 0; sn < 2; ++sn) {
      int gn = bn + wn + sn * 16 + nloc;
      if (gn >= N) continue;
      float bv = bias ? bias[gn] : 0.f;
      int gm0 = bm + wm + sm * 16 + mhalf;
#pragma unroll
      for (int vg = 0; vg < 8; ++vg) {
        int gm = gm0 + vg;
        if (gm < M) {
          float val = acc[sm][sn].f[vg] + bv;
          if (Cf) Cf[coff + (long)gm * ldc + gn] = val;
          if (Cb) Cb[coff + (long)gm * ldc + gn] = f32_to_bf16_bits(val);
        }
      }
    }
}

// ---------------- attention scores: out[bh][u][s] = q_row(u) . k_row(s), per head ----------------
// K=64 is contiguous per head in both operands, so fragments load straight from global.
__global__ __launch_bounds__(256) void scores_kernel(
    const unsigned short* __restrict__ qb,   // [B][QROWS][Dm] bf16 (projected)
    const unsigned short* __restrict__ kp,   // [B][Sk][Dm]   bf16 (projected)
    const int* __restrict__ mtop,            // null (sample: rows 0..44) or [B*H*45] (final: rows 45+idx)
    float* __restrict__ out) {               // [B*H][Uu][Sk]
  int bh = blockIdx.z;
  int b = bh / Hh, h = bh % Hh;
  int m0 = blockIdx.y * 16;
  int lane = threadIdx.x & 31, wid = threadIdx.x >> 5;
  int s0 = blockIdx.x * 128 + wid * 16;

  int u  = m0 + (lane & 15);
  int uc = (u < Uu) ? u : (Uu - 1);
  int arow = mtop ? (Uu + mtop[bh * Uu + uc]) : uc;
  const unsigned short* arp = qb + ((long)b * QROWS + arow) * Dm + h * Dk;
  const unsigned short* brp = kp + ((long)b * Sk + s0 + (lane & 15)) * Dm + h * Dk;

  FragC acc;
#pragma unroll
  for (int e = 0; e < 8; ++e) acc.f[e] = 0.f;
#pragma unroll
  for (int k0 = 0; k0 < Dk; k0 += 32) {
    v16bf af = load_frag_a(arp + k0, lane);
    v16bf bf = load_frag_b(brp + k0, lane);
    acc.v = __builtin_amdgcn_wmma_f32_16x16x32_bf16(false, af, false, bf, (short)0, acc.v, false, false);
  }
  int nloc = lane & 15, mhalf = (lane >> 4) * 8;
#pragma unroll
  for (int vg = 0; vg < 8; ++vg) {
    int um = m0 + mhalf + vg;
    if (um < Uu) out[((long)bh * Uu + um) * Sk + s0 + nloc] = acc.f[vg];
  }
}

// ---------------- M = rowmax - rowmean over S ----------------
__global__ __launch_bounds__(256) void reduce_m_kernel(const float* __restrict__ scores,
                                                       float* __restrict__ Mv) {
  __shared__ float smx[256], ssm[256];
  int row = blockIdx.x;  // B*H*Uu
  const float* p = scores + (long)row * Sk;
  float mx = -3.4e38f, sm = 0.f;
  for (int i = threadIdx.x; i < Sk; i += 256) { float v = p[i]; mx = fmaxf(mx, v); sm += v; }
  smx[threadIdx.x] = mx; ssm[threadIdx.x] = sm;
  __syncthreads();
  for (int s = 128; s > 0; s >>= 1) {
    if (threadIdx.x < s) {
      smx[threadIdx.x] = fmaxf(smx[threadIdx.x], smx[threadIdx.x + s]);
      ssm[threadIdx.x] += ssm[threadIdx.x + s];
    }
    __syncthreads();
  }
  if (threadIdx.x == 0) Mv[row] = smx[0] - ssm[0] / (float)Sk;
}

// ---------------- top-k over 45 values = rank sort (stable, descending) ----------------
__global__ void topk_kernel(const float* __restrict__ Mv, int* __restrict__ mtop) {
  __shared__ float vals[64];
  int bh = blockIdx.x, i = threadIdx.x;
  if (i < Uu) vals[i] = Mv[bh * Uu + i];
  __syncthreads();
  if (i < Uu) {
    float v = vals[i]; int r = 0;
    for (int j = 0; j < Uu; ++j) {
      float w = vals[j];
      if (w > v || (w == v && j < i)) ++r;
    }
    mtop[bh * Uu + r] = i;
  }
}

// ---------------- softmax row: f32 attn to d_out, bf16 copy for context GEMM ----------------
__global__ __launch_bounds__(256) void softmax_kernel(const float* __restrict__ scores,
                                                      float* __restrict__ attn,
                                                      unsigned short* __restrict__ attnb) {
  __shared__ float red[256];
  int row = blockIdx.x;  // B*H*Uu
  const float* p = scores + (long)row * Sk;
  float mx = -3.4e38f;
  for (int i = threadIdx.x; i < Sk; i += 256) mx = fmaxf(mx, p[i]);
  red[threadIdx.x] = mx; __syncthreads();
  for (int s = 128; s > 0; s >>= 1) {
    if (threadIdx.x < s) red[threadIdx.x] = fmaxf(red[threadIdx.x], red[threadIdx.x + s]);
    __syncthreads();
  }
  float rmax = red[0]; __syncthreads();
  float sm = 0.f;
  for (int i = threadIdx.x; i < Sk; i += 256) sm += __expf(p[i] - rmax);
  red[threadIdx.x] = sm; __syncthreads();
  for (int s = 128; s > 0; s >>= 1) {
    if (threadIdx.x < s) red[threadIdx.x] += red[threadIdx.x + s];
    __syncthreads();
  }
  float inv = 1.f / red[0];
  for (int i = threadIdx.x; i < Sk; i += 256) {
    float a = __expf(p[i] - rmax) * inv;
    attn[(long)row * Sk + i]  = a;
    attnb[(long)row * Sk + i] = f32_to_bf16_bits(a);
  }
}

// ---------------- host-side orchestration ----------------
static inline size_t alignup(size_t x) { return (x + 255) & ~(size_t)255; }

extern "C" void kernel_launch(void* const* d_in, const int* in_sizes, int n_in,
                              void* d_out, int out_size, void* d_ws, size_t ws_size,
                              hipStream_t stream) {
  const float* queries = (const float*)d_in[0];
  const float* keys    = (const float*)d_in[1];
  const float* values  = (const float*)d_in[2];
  const int*   perm    = (const int*)d_in[3];
  const float* Wq = (const float*)d_in[4];  const float* bq = (const float*)d_in[5];
  const float* Wk = (const float*)d_in[6];  const float* bk = (const float*)d_in[7];
  const float* Wv = (const float*)d_in[8];  const float* bv = (const float*)d_in[9];
  const float* Wo = (const float*)d_in[10]; const float* bo = (const float*)d_in[11];
  (void)in_sizes; (void)n_in; (void)out_size; (void)ws_size;

  char* w = (char*)d_ws;
  size_t off = 0;
  auto alloc = [&](size_t bytes) { void* p = w + off; off = alignup(off + bytes); return p; };

  unsigned short* wq_b = (unsigned short*)alloc((size_t)Dm * Dm * 2);
  unsigned short* wk_b = (unsigned short*)alloc((size_t)Dm * Dm * 2);
  unsigned short* wv_b = (unsigned short*)alloc((size_t)Dm * Dm * 2);
  unsigned short* wo_b = (unsigned short*)alloc((size_t)Dm * Dm * 2);
  unsigned short* keys_b   = (unsigned short*)alloc((size_t)B_ * Sk * Dm * 2);
  unsigned short* values_b = (unsigned short*)alloc((size_t)B_ * Sk * Dm * 2);
  unsigned short* qg_b = (unsigned short*)alloc((size_t)B_ * QROWS * Dm * 2);
  unsigned short* qp_b = (unsigned short*)alloc((size_t)B_ * QROWS * Dm * 2);
  unsigned short* kp_b = (unsigned short*)alloc((size_t)B_ * Sk * Dm * 2);
  unsigned short* vp_b = (unsigned short*)alloc((size_t)B_ * Sk * Dm * 2);
  float*          sc_f = (float*)alloc((size_t)B_ * Hh * Uu * Sk * 4);
  unsigned short* at_b = (unsigned short*)alloc((size_t)B_ * Hh * Uu * Sk * 2);
  float*          Mv   = (float*)alloc((size_t)B_ * Hh * Uu * 4);
  int*            mtop = (int*)alloc((size_t)B_ * Hh * Uu * 4);
  float*          ctxf = (float*)alloc((size_t)B_ * Uu * Dm * 4);
  unsigned short* ctxb = (unsigned short*)alloc((size_t)B_ * Uu * Dm * 2);

  float* out_proj = (float*)d_out;                       // (B, 45, 1024)
  float* attn_out = (float*)d_out + (long)B_ * Uu * Dm;  // (B, H, 45, S)

  // 1) bf16 conversions
  cvt_bf16_kernel<<<1024, 256, 0, stream>>>(Wq, wq_b, (long)Dm * Dm);
  cvt_bf16_kernel<<<1024, 256, 0, stream>>>(Wk, wk_b, (long)Dm * Dm);
  cvt_bf16_kernel<<<1024, 256, 0, stream>>>(Wv, wv_b, (long)Dm * Dm);
  cvt_bf16_kernel<<<1024, 256, 0, stream>>>(Wo, wo_b, (long)Dm * Dm);
  cvt_bf16_kernel<<<4096, 256, 0, stream>>>(keys,   keys_b,   (long)B_ * Sk * Dm);
  cvt_bf16_kernel<<<4096, 256, 0, stream>>>(values, values_b, (long)B_ * Sk * Dm);
  gather_q_kernel<<<B_ * QROWS, 256, 0, stream>>>(queries, perm, qg_b);

  // 2) projections (k, v full; q only the 90 rows that matter)
  gemm_bf16_kernel<<<dim3(Dm / BN, Sk / BM, B_), 256, 0, stream>>>(
      keys_b, wk_b, bk, nullptr, kp_b, Sk, Dm, Dm, Dm, Dm, Dm, 1,
      (long)Sk * Dm, 0, 0, 0, (long)Sk * Dm, 0);
  gemm_bf16_kernel<<<dim3(Dm / BN, Sk / BM, B_), 256, 0, stream>>>(
      values_b, wv_b, bv, nullptr, vp_b, Sk, Dm, Dm, Dm, Dm, Dm, 1,
      (long)Sk * Dm, 0, 0, 0, (long)Sk * Dm, 0);
  gemm_bf16_kernel<<<dim3(Dm / BN, 1, B_), 256, 0, stream>>>(
      qg_b, wq_b, bq, nullptr, qp_b, QROWS, Dm, Dm, Dm, Dm, Dm, 1,
      (long)QROWS * Dm, 0, 0, 0, (long)QROWS * Dm, 0);

  // 3) sample scores -> M -> top-k
  scores_kernel<<<dim3(Sk / 128, (Uu + 15) / 16, B_ * Hh), 256, 0, stream>>>(
      qp_b, kp_b, nullptr, sc_f);
  reduce_m_kernel<<<B_ * Hh * Uu, 256, 0, stream>>>(sc_f, Mv);
  topk_kernel<<<B_ * Hh, 64, 0, stream>>>(Mv, mtop);

  // 4) final scores with q_red indirection, then softmax
  scores_kernel<<<dim3(Sk / 128, (Uu + 15) / 16, B_ * Hh), 256, 0, stream>>>(
      qp_b, kp_b, mtop, sc_f);
  softmax_kernel<<<B_ * Hh * Uu, 256, 0, stream>>>(sc_f, attn_out, at_b);

  // 5) context = attn @ v (per head; head-strided B/C addressing), then output projection
  gemm_bf16_kernel<<<dim3(1, 1, B_ * Hh), 256, 0, stream>>>(
      at_b, vp_b, nullptr, ctxf, ctxb, Uu, Dk, Sk, Sk, Dm, Dm, Hh,
      (long)Hh * Uu * Sk, (long)Uu * Sk,   // A: attn per (b,h)
      (long)Sk * Dm, (long)Dk,             // B: v[b, :, h*64 + n]
      (long)Uu * Dm, (long)Dk);            // C: ctx[b, u, h*64 + n]
  gemm_bf16_kernel<<<dim3(Dm / BN, (B_ * Uu + BM - 1) / BM, 1), 256, 0, stream>>>(
      ctxb, wo_b, bo, out_proj, nullptr, B_ * Uu, Dm, Dm, Dm, Dm, Dm, 1,
      0, 0, 0, 0, 0, 0);
}
